// TransformerBlock_1176821039971
// MI455X (gfx1250) — matrix operand order repaired and checked by reading, not compiled, not run on hardware
//
#include <hip/hip_runtime.h>
#include <hip/hip_bf16.h>
#include <cstdint>

#define BB   4
#define SS   1024
#define DD   768
#define HH   12
#define HDIM 64
#define FFD  3072

typedef _Float16 half_t;
typedef __attribute__((ext_vector_type(16))) _Float16 v16h;
typedef __attribute__((ext_vector_type(8)))  _Float16 h8;
typedef __attribute__((ext_vector_type(8)))  float    v8f;

// ---------------------------------------------------------------------------
// WMMA fragment helpers (CDNA5 wave32 layouts, ISA 7.12.2)
// A: 16x32 f16 (MxK). lane&15 = M row; lanes 16..31 take K+8; elems 8..15 = K+16.
// ---------------------------------------------------------------------------
static __device__ __forceinline__ v16h load_a_frag(const half_t* base, int ld, int lane) {
  const int m  = lane & 15;
  const int kb = (lane >> 4) << 3;
  const h8 lo = *(const h8*)(base + (size_t)m * ld + kb);
  const h8 hi = *(const h8*)(base + (size_t)m * ld + kb + 16);
  v16h r;
#pragma unroll
  for (int i = 0; i < 8; ++i) { r[i] = lo[i]; r[8 + i] = hi[i]; }
  return r;
}

// B: 32x16 f16 (KxN) where B[k][n] = W[n][k], W row-major with leading dim ld.
// lane&15 = N col; lanes 16..31 take K+8; elems 8..15 = K+16 (same as A).
static __device__ __forceinline__ v16h load_b_frag(const half_t* base, int ld, int lane) {
  const int n  = lane & 15;
  const int kb = (lane >> 4) << 3;
  const h8 lo = *(const h8*)(base + (size_t)n * ld + kb);
  const h8 hi = *(const h8*)(base + (size_t)n * ld + kb + 16);
  v16h r;
#pragma unroll
  for (int i = 0; i < 8; ++i) { r[i] = lo[i]; r[8 + i] = hi[i]; }
  return r;
}

static __device__ __forceinline__ v8f wmma_f16(v16h a, v16h b, v8f c) {
  return __builtin_amdgcn_wmma_f32_16x16x32_f16(false, a, false, b, (short)0, c, false, false);
}

static __device__ __forceinline__ float gelu_tanh(float u) {
  return 0.5f * u * (1.0f + tanhf(0.7978845608028654f * (u + 0.044715f * u * u * u)));
}

// Async global->LDS lane copy (16B per lane), tracked by ASYNCcnt (ISA 10.7).
// VDST = per-lane LDS byte offset, VADDR = per-lane 64-bit global address.
static __device__ __forceinline__ void async_load_b128(unsigned lds_off, const void* gptr) {
  asm volatile("global_load_async_to_lds_b128 %0, %1, off"
               :: "v"(lds_off), "v"((unsigned long long)(uintptr_t)gptr)
               : "memory");
}
static __device__ __forceinline__ void wait_async0() {
  asm volatile("s_wait_asynccnt 0" ::: "memory");
}

// ---------------------------------------------------------------------------
// GEMM: C[M,N] = A[M,K] @ W[N,K]^T, f16 inputs, f32 accumulate.
// Block = 8 waves = 512x64 macro-tile; the 64xK weight tile is shared by all
// waves and streamed through LDS with double-buffered async global->LDS
// copies (one GLOBAL_LOAD_ASYNC_TO_LDS_B128 per thread per K-step).
// Epilogue: optional bias (f32), GELU, residual (f32), f32 and/or f16 output.
// Grids are exact: M % 512 == 0, N % 64 == 0 (no early-out waves; EXEC full).
// ---------------------------------------------------------------------------
__global__ __launch_bounds__(256) void gemm_ws(
    const half_t* __restrict__ A, const half_t* __restrict__ W,
    const float* __restrict__ bias, const float* __restrict__ resid,
    float* __restrict__ outF, half_t* __restrict__ outH,
    int M, int N, int Kd, int act)
{
  __shared__ __align__(16) half_t bsh[2][64 * 32];   // 2 x 4KB double buffer

  const int lane = threadIdx.x & 31;
  const int wid  = threadIdx.x >> 5;
  const int nblk = N >> 6;
  const int bm   = blockIdx.x / nblk;
  const int bn   = blockIdx.x % nblk;
  const int row0 = bm * 512 + wid * 64;
  const int col0 = bn * 64;

  // async staging: 64 rows x 32 halfs = 256 x 16B, one 16B lane-load per thread
  const int idx = (wid << 5) | lane;        // 0..255
  const int sn  = idx >> 2;                 // weight row within tile (0..63)
  const int sq  = idx & 3;                  // 8-half quarter of the 32-half row
  const unsigned lb0 = (unsigned)(uintptr_t)(&bsh[0][0]);
  const unsigned lb1 = (unsigned)(uintptr_t)(&bsh[1][0]);
  const unsigned loff = (unsigned)(sn * 64 + sq * 16);          // bytes in buffer
  const half_t* gsrc = W + (size_t)(col0 + sn) * Kd + sq * 8;   // row base at k=0

  // prologue: stage k0 = 0 into buffer 0
  async_load_b128(lb0 + loff, gsrc);
  wait_async0();
  __syncthreads();

  v8f acc[4][4] = {};
  const int steps = Kd >> 5;
  for (int t = 0; t < steps; ++t) {
    const int k0 = t << 5;
    if (t + 1 < steps)                        // prefetch next weight chunk to LDS
      async_load_b128(((t & 1) ? lb0 : lb1) + loff, gsrc + k0 + 32);

    const half_t* bcur = &bsh[t & 1][0];
    v16h af[4], bf[4];
#pragma unroll
    for (int i = 0; i < 4; ++i)
      af[i] = load_a_frag(A + (size_t)(row0 + 16 * i) * Kd + k0, Kd, lane);
#pragma unroll
    for (int j = 0; j < 4; ++j)
      bf[j] = load_b_frag(bcur + (16 * j) * 32, 32, lane);
    if (k0 + 32 < Kd)                          // global_prefetch_b8 on the A stream
      __builtin_prefetch(A + (size_t)(row0 + lane) * Kd + k0 + 32, 0, 1);

#pragma unroll
    for (int i = 0; i < 4; ++i)
#pragma unroll
      for (int j = 0; j < 4; ++j)
        acc[i][j] = wmma_f16(af[i], bf[j], acc[i][j]);

    wait_async0();                             // own wave's async copy done
    __syncthreads();                           // all waves done reading bcur
  }

  const int n_in = lane & 15;
  const int mb   = (lane >> 4) << 3;
#pragma unroll
  for (int i = 0; i < 4; ++i) {
#pragma unroll
    for (int j = 0; j < 4; ++j) {
      const int ncol = col0 + 16 * j + n_in;
#pragma unroll
      for (int r = 0; r < 8; ++r) {
        const int mrow = row0 + 16 * i + mb + r;
        float v = acc[i][j][r];
        if (bias)     v += bias[ncol];
        if (act == 1) v  = gelu_tanh(v);
        if (resid)    v += resid[(size_t)mrow * N + ncol];
        if (outF) outF[(size_t)mrow * N + ncol] = v;
        if (outH) outH[(size_t)mrow * N + ncol] = (half_t)v;
      }
    }
  }
}

// ---------------------------------------------------------------------------
// LayerNorm over D=768 (one block of 256 threads per row), f32 in -> f16 out.
// ---------------------------------------------------------------------------
__global__ __launch_bounds__(256) void layernorm_to_f16(
    const float* __restrict__ x, const float* __restrict__ g,
    const float* __restrict__ sh, half_t* __restrict__ out)
{
  __shared__ float red[256];
  const int row = blockIdx.x;
  const int t   = threadIdx.x;
  const float* xr = x + (size_t)row * DD;
  float v0 = xr[t], v1 = xr[t + 256], v2 = xr[t + 512];
  red[t] = v0 + v1 + v2;
  __syncthreads();
  for (int s = 128; s > 0; s >>= 1) { if (t < s) red[t] += red[t + s]; __syncthreads(); }
  const float mean = red[0] / (float)DD;
  __syncthreads();
  float d0 = v0 - mean, d1 = v1 - mean, d2 = v2 - mean;
  red[t] = d0 * d0 + d1 * d1 + d2 * d2;
  __syncthreads();
  for (int s = 128; s > 0; s >>= 1) { if (t < s) red[t] += red[t + s]; __syncthreads(); }
  const float rstd = rsqrtf(red[0] / (float)DD + 1e-5f);
  half_t* orow = out + (size_t)row * DD;
  orow[t]       = (half_t)(d0 * rstd * g[t]       + sh[t]);
  orow[t + 256] = (half_t)(d1 * rstd * g[t + 256] + sh[t + 256]);
  orow[t + 512] = (half_t)(d2 * rstd * g[t + 512] + sh[t + 512]);
}

// ---------------------------------------------------------------------------
// Weight conversion f32 -> f16.
// ---------------------------------------------------------------------------
__global__ void cvt_f32_f16(const float* __restrict__ in, half_t* __restrict__ out, int n) {
  int i = blockIdx.x * blockDim.x + threadIdx.x;
  if (i < n) out[i] = (half_t)in[i];
}

// Per-head transpose of V: flat index b*S*D + h*S*HD + s*HD + d  ->  Vt[(b*H+h)*HD + d][s]
__global__ void transpose_v(const half_t* __restrict__ V, half_t* __restrict__ Vt) {
  int i = blockIdx.x * blockDim.x + threadIdx.x;
  if (i >= BB * SS * DD) return;
  const int d  = i & 63;
  const int s  = (i >> 6) & 1023;
  const int bh = i >> 16;                 // S*HD == 65536
  Vt[((size_t)bh * HDIM + d) * SS + s] = V[i];
}

// ---------------------------------------------------------------------------
// Causal attention, one wave per (b,h,16 query rows).
// The reference's head split is a *plain view*, so each head's Q/K is a
// contiguous (S,HD) matrix at offset b*S*D + h*S*HD.
// Scores (16x1024 f32) in LDS; probabilities overwrite the same rows as f16
// (writes only clobber already-consumed earlier bytes; volatile keeps order).
// ---------------------------------------------------------------------------
__global__ __launch_bounds__(32) void attn_ws(
    const half_t* __restrict__ Q, const half_t* __restrict__ Kc,
    const half_t* __restrict__ Vt, half_t* __restrict__ O)
{
  extern __shared__ float sc[];            // 16 * 1024 f32 = 64 KB
  const int lane = threadIdx.x;
  const int nqt  = SS / 16;
  const int bh   = blockIdx.x / nqt;
  const int qt   = blockIdx.x % nqt;
  const int b    = bh / HH;
  const int h    = bh % HH;
  const int q0   = qt * 16;
  const half_t* Qh = Q  + (size_t)b * SS * DD + (size_t)h * SS * HDIM;
  const half_t* Kh = Kc + (size_t)b * SS * DD + (size_t)h * SS * HDIM;
  const half_t* Vh = Vt + (size_t)bh * HDIM * SS;

  const int n_in = lane & 15;
  const int mb   = (lane >> 4) << 3;

  // Q fragments for this 16-row tile, contraction over HD=64 (two K chunks)
  v16h qa0 = load_a_frag(Qh + (size_t)q0 * HDIM + 0,  HDIM, lane);
  v16h qa1 = load_a_frag(Qh + (size_t)q0 * HDIM + 32, HDIM, lane);

  // scores = Q K^T / HD with causal mask; only tiles with k0 <= q0+15
  for (int kt = 0; kt <= qt; ++kt) {
    const int k0 = kt * 16;
    v16h kb0 = load_b_frag(Kh + (size_t)k0 * HDIM + 0,  HDIM, lane);
    v16h kb1 = load_b_frag(Kh + (size_t)k0 * HDIM + 32, HDIM, lane);
    v8f s = {};
    s = wmma_f16(qa0, kb0, s);
    s = wmma_f16(qa1, kb1, s);
#pragma unroll
    for (int r = 0; r < 8; ++r) {
      const int m   = mb + r;
      const int key = k0 + n_in;
      float v = (key <= q0 + m) ? s[r] * (1.0f / 64.0f) : -3.0e38f;
      sc[m * 1024 + key] = v;
    }
  }
  __syncthreads();

  const int kend = ((q0 + 16 + 31) >> 5) << 5;   // keys padded to 32, <= 1024
  {
    // rows handled by lane&15 (lanes 16..31 duplicate -> EXEC stays full)
    const int row = lane & 15;
    const int len = q0 + row + 1;
    volatile float* srow = sc + row * 1024;
    float mx = -3.0e38f;
    for (int j = 0; j < len; ++j) mx = fmaxf(mx, srow[j]);
    float sum = 0.f;
    for (int j = 0; j < len; ++j) sum += __expf(srow[j] - mx);
    const float inv = 1.0f / sum;
    volatile half_t* prow = (volatile half_t*)(sc + row * 1024);  // same row base
    for (int j = 0; j < kend; ++j) {            // uniform trip count
      float p = 0.f;
      if (j < len) p = __expf(srow[j] - mx) * inv;
      prow[j] = (half_t)p;                      // clobbers only f32[j/2] (< j): safe
    }
  }
  __syncthreads();

  // out = P @ V, P in LDS as f16 with row stride 2048 halves (4096B == f32 row)
  const half_t* P = (const half_t*)sc;
  v8f oacc[4] = {};
  for (int kp = 0; kp < kend; kp += 32) {
    v16h pa = load_a_frag(P + kp, 2048, lane);
#pragma unroll
    for (int nf = 0; nf < 4; ++nf) {
      v16h vb = load_b_frag(Vh + (size_t)(nf * 16) * SS + kp, SS, lane);
      oacc[nf] = wmma_f16(pa, vb, oacc[nf]);
    }
  }
#pragma unroll
  for (int nf = 0; nf < 4; ++nf)
#pragma unroll
    for (int r = 0; r < 8; ++r) {
      const int m   = mb + r;
      const int col = h * HDIM + nf * 16 + n_in;   // merged (swapaxes) layout
      O[((size_t)b * SS + q0 + m) * DD + col] = (half_t)oacc[nf][r];
    }
}

// ---------------------------------------------------------------------------
// Host-side orchestration.
// ---------------------------------------------------------------------------
extern "C" void kernel_launch(void* const* d_in, const int* in_sizes, int n_in,
                              void* d_out, int out_size, void* d_ws, size_t ws_size,
                              hipStream_t stream) {
  const float* x  = (const float*)d_in[0];
  const float* Wq = (const float*)d_in[1];
  const float* Wk = (const float*)d_in[2];
  const float* Wv = (const float*)d_in[3];
  const float* Wo = (const float*)d_in[4];
  const float* W1 = (const float*)d_in[5];
  const float* b1 = (const float*)d_in[6];
  const float* W2 = (const float*)d_in[7];
  const float* b2 = (const float*)d_in[8];
  const float* g1 = (const float*)d_in[9];
  const float* s1 = (const float*)d_in[10];
  const float* g2 = (const float*)d_in[11];
  const float* s2 = (const float*)d_in[12];

  // workspace layout (bytes)
  unsigned char* ws = (unsigned char*)d_ws;
  constexpr size_t SZ_ACT  = (size_t)BB * SS * DD * 2;     // 6,291,456  (f16 B*S*D)
  constexpr size_t SZ_DXD  = (size_t)DD * DD * 2;          // 1,179,648
  constexpr size_t SZ_FW   = (size_t)FFD * DD * 2;         // 4,718,592
  constexpr size_t O_H1 = 0;
  constexpr size_t O_WQ = O_H1 + SZ_ACT;
  constexpr size_t O_WK = O_WQ + SZ_DXD;
  constexpr size_t O_WV = O_WK + SZ_DXD;
  constexpr size_t O_WO = O_WV + SZ_DXD;
  constexpr size_t O_W1 = O_WO + SZ_DXD;
  constexpr size_t O_W2 = O_W1 + SZ_FW;
  constexpr size_t O_QF = O_W2 + SZ_FW;
  constexpr size_t O_KF = O_QF + SZ_ACT;
  constexpr size_t O_VF = O_KF + SZ_ACT;
  constexpr size_t O_VT = O_VF + SZ_ACT;
  constexpr size_t O_AO = O_VT + SZ_ACT;
  constexpr size_t O_X1 = O_AO + SZ_ACT;
  constexpr size_t O_H2 = O_X1 + (size_t)BB * SS * DD * 4; // x1 f32
  constexpr size_t O_FF = O_H2 + SZ_ACT;

  half_t* h1   = (half_t*)(ws + O_H1);
  half_t* wq16 = (half_t*)(ws + O_WQ);
  half_t* wk16 = (half_t*)(ws + O_WK);
  half_t* wv16 = (half_t*)(ws + O_WV);
  half_t* wo16 = (half_t*)(ws + O_WO);
  half_t* w116 = (half_t*)(ws + O_W1);
  half_t* w216 = (half_t*)(ws + O_W2);
  half_t* qf   = (half_t*)(ws + O_QF);
  half_t* kf   = (half_t*)(ws + O_KF);
  half_t* vf   = (half_t*)(ws + O_VF);
  half_t* vt   = (half_t*)(ws + O_VT);
  half_t* ao   = (half_t*)(ws + O_AO);
  float*  x1   = (float*)(ws + O_X1);
  half_t* h2   = (half_t*)(ws + O_H2);
  half_t* ffb  = (half_t*)(ws + O_FF);

  auto cvt = [&](const float* src, half_t* dst, int n) {
    cvt_f32_f16<<<(n + 255) / 256, 256, 0, stream>>>(src, dst, n);
  };
  cvt(Wq, wq16, DD * DD);
  cvt(Wk, wk16, DD * DD);
  cvt(Wv, wv16, DD * DD);
  cvt(Wo, wo16, DD * DD);
  cvt(W1, w116, FFD * DD);
  cvt(W2, w216, DD * FFD);

  const int M = BB * SS;                      // 4096
  // grid = (M/512) * (N/64) blocks of 8 waves
  const int gDxD = (M / 512) * (DD / 64);     // 96
  const int gFF  = (M / 512) * (FFD / 64);    // 384

  // h1 = LN(x)
  layernorm_to_f16<<<M, 256, 0, stream>>>(x, g1, s1, h1);
  // Q, K, V projections (f16 outputs)
  gemm_ws<<<gDxD, 256, 0, stream>>>(h1, wq16, nullptr, nullptr, nullptr, qf, M, DD, DD, 0);
  gemm_ws<<<gDxD, 256, 0, stream>>>(h1, wk16, nullptr, nullptr, nullptr, kf, M, DD, DD, 0);
  gemm_ws<<<gDxD, 256, 0, stream>>>(h1, wv16, nullptr, nullptr, nullptr, vf, M, DD, DD, 0);
  // per-head V transpose
  transpose_v<<<(BB * SS * DD + 255) / 256, 256, 0, stream>>>(vf, vt);
  // causal attention
  attn_ws<<<BB * HH * (SS / 16), 32, 16 * 1024 * sizeof(float), stream>>>(qf, kf, vt, ao);
  // x1 = x + attn @ Wo^T
  gemm_ws<<<gDxD, 256, 0, stream>>>(ao, wo16, nullptr, x, x1, nullptr, M, DD, DD, 0);
  // h2 = LN(x1)
  layernorm_to_f16<<<M, 256, 0, stream>>>(x1, g2, s2, h2);
  // ff = gelu(h2 @ W1^T + b1)
  gemm_ws<<<gFF, 256, 0, stream>>>(h2, w116, b1, nullptr, nullptr, ffb, M, FFD, DD, 1);
  // out = x1 + ff @ W2^T + b2
  gemm_ws<<<gDxD, 256, 0, stream>>>(ffb, w216, b2, x1, (float*)d_out, nullptr, M, DD, FFD, 0);
}
